// Encoder_5892695130542
// MI455X (gfx1250) — matrix operand order repaired and checked by reading, not compiled, not run on hardware
//
#include <hip/hip_runtime.h>
#include <hip/hip_bf16.h>
#include <stdint.h>

// Problem constants (match reference)
#define B_   8
#define T_   512
#define C_   768
#define FC_  3072
#define H_   12
#define L_   6
#define DK_  64
#define TP_  514          // T + 2 (zero pad column on each side for K=3 conv)
#define WIN_ 4
#define SCALE_ 0.125f     // 1/sqrt(64)

typedef unsigned short u16;
typedef __attribute__((ext_vector_type(16))) __bf16 v16bf;
typedef __attribute__((ext_vector_type(8)))  float  v8f;
typedef int gv4i __attribute__((vector_size(16)));   // matches builtin prototype (V4i)

struct __align__(16) U4 { unsigned a, b, c, d; };
union FragBF { U4 q[2]; v16bf v; };

// CDNA5 async global->LDS DMA (ASYNCcnt-tracked), with graceful fallback.
#if defined(__HIP_DEVICE_COMPILE__) && defined(__gfx1250__) && \
    __has_builtin(__builtin_amdgcn_global_load_async_to_lds_b128)
#define HAVE_ASYNC_LDS 1
#else
#define HAVE_ASYNC_LDS 0
#endif

#define AS1 __attribute__((address_space(1)))
#define AS3 __attribute__((address_space(3)))

__device__ __forceinline__ void wait_async_all() {
#if HAVE_ASYNC_LDS
#if __has_builtin(__builtin_amdgcn_s_wait_asynccnt)
  __builtin_amdgcn_s_wait_asynccnt(0);
#else
  asm volatile("s_wait_asynccnt 0x0" ::: "memory");
#endif
#endif
}

__device__ __forceinline__ u16 f2b(float f) {
  unsigned u = __float_as_uint(f);
  u += 0x7FFFu + ((u >> 16) & 1u);          // round-to-nearest-even
  return (u16)(u >> 16);
}
__device__ __forceinline__ float b2f(u16 h) {
  return __uint_as_float(((unsigned)h) << 16);
}

// ---------------------------------------------------------------------------
// Generic bf16 WMMA GEMM:  C[z][M][N] = alpha * sum_kb A_kb[M x K] * B_kb[K x N]
//   TA==0: A[m,k] at A[m*lda + k]      TA==1: A stored [K x M]: A[k*lda + m]
//   TB==0: B[k,n] at B[k*ldb + n]      TB==1: B stored [N x K]: B[n*ldb + k]
//   Epilogue: +bias[m], ReLU, *maskN[z*mZ + n], f32 or bf16 output at
//             cBase + z*sCz + m*sm + n*sn
// Block: 256 threads = 8 waves; block tile 128x64; wave tile 16x64 (4 wmma/kstep)
// TA==0 / TB==1 operands are 16B-aligned by construction -> async b128 DMA to LDS.
// ---------------------------------------------------------------------------
template <int TA, int TB>
__global__ __launch_bounds__(256) void gemm_bf16_wmma(
    const u16* __restrict__ A, long long lda, long long sAz, long long sAkb,
    const u16* __restrict__ Bm, long long ldb, long long sBz, long long sBkb,
    void* __restrict__ Cv, long long cBase, long long sCz, long long sm, long long sn,
    int outBf, const float* __restrict__ bias, const float* __restrict__ maskN,
    long long mZ, int Kd, int nkb, float alpha, int doRelu)
{
  __shared__ alignas(16) u16 sA[128 * 40];   // [m][k], row stride 40 u16 (80B)
  __shared__ alignas(16) u16 sB[64 * 40];    // transposed: [n][k]

  const int tid  = threadIdx.x;
  const int lane = tid & 31;
  const int wave = tid >> 5;
  const int z    = blockIdx.z;
  const long long gm0 = (long long)blockIdx.x * 128;
  const long long gn0 = (long long)blockIdx.y * 64;

  const u16* Az = A  + (long long)z * sAz;
  const u16* Bz = Bm + (long long)z * sBz;

  v8f acc[4];
  #pragma unroll
  for (int i = 0; i < 4; ++i)
    #pragma unroll
    for (int j = 0; j < 8; ++j) acc[i][j] = 0.0f;

  const int r  = lane & 15;
  const int kh = lane >> 4;

  for (int kb = 0; kb < nkb; ++kb) {
    const u16* Ak = Az + (long long)kb * sAkb;
    const u16* Bk = Bz + (long long)kb * sBkb;
    for (int kk = 0; kk < Kd; kk += 32) {
      __syncthreads();
      // ---- stage A tile 128x32 ----
#if HAVE_ASYNC_LDS
      if (TA == 0) {
        // 512 16B chunks, 2 per thread; global & LDS both 16B aligned.
        #pragma unroll
        for (int ii = 0; ii < 2; ++ii) {
          int i = tid + ii * 256;
          int m = i >> 2, k0 = (i & 3) * 8;
          const u16* gp = Ak + (long long)(gm0 + m) * lda + (kk + k0);
          __builtin_amdgcn_global_load_async_to_lds_b128(
              (AS1 gv4i*)(gp), (AS3 gv4i*)(&sA[m * 40 + k0]), 0, 0);
        }
      } else
#endif
      {
        #pragma unroll
        for (int i = tid; i < 128 * 32; i += 256) {
          int m = i >> 5, k = i & 31;
          long long gi = TA ? ((long long)(kk + k) * lda + (gm0 + m))
                            : ((long long)(gm0 + m) * lda + (kk + k));
          sA[m * 40 + k] = Ak[gi];
        }
      }
      // ---- stage B tile 32x64 into transposed LDS [n][k] ----
#if HAVE_ASYNC_LDS
      if (TB == 1) {
        // global rows are [n][k]: contiguous along k, same as LDS layout.
        int n = tid >> 2, k0 = (tid & 3) * 8;
        const u16* gp = Bk + (long long)(gn0 + n) * ldb + (kk + k0);
        __builtin_amdgcn_global_load_async_to_lds_b128(
            (AS1 gv4i*)(gp), (AS3 gv4i*)(&sB[n * 40 + k0]), 0, 0);
      } else
#endif
      {
        #pragma unroll
        for (int i = tid; i < 32 * 64; i += 256) {
          int k = i >> 6, n = i & 63;
          long long gi = TB ? ((long long)(gn0 + n) * ldb + (kk + k))
                            : ((long long)(kk + k) * ldb + (gn0 + n));
          sB[n * 40 + k] = Bk[gi];
        }
      }
      wait_async_all();
      __syncthreads();

      // A fragment (16x32 bf16): lane r=row; lanes<16 take K 0-7 & 16-23,
      // lanes>=16 take K 8-15 & 24-31 (ISA 7.12.2), as two b128 LDS reads.
      const U4* ar = reinterpret_cast<const U4*>(&sA[(wave * 16 + r) * 40]);
      FragBF fa; fa.q[0] = ar[kh]; fa.q[1] = ar[2 + kh];
      #pragma unroll
      for (int nn = 0; nn < 4; ++nn) {
        // B fragment (32x16): lane r=col; lanes<16 K 0-7 & 16-23, lanes>=16 K 8-15 & 24-31.
        const U4* br = reinterpret_cast<const U4*>(&sB[(nn * 16 + r) * 40]);
        FragBF fb; fb.q[0] = br[kh]; fb.q[1] = br[2 + kh];
        acc[nn] = __builtin_amdgcn_wmma_f32_16x16x32_bf16(
            false, fa.v, false, fb.v, (short)0, acc[nn], false, false);
      }
    }
  }

  // Epilogue. C frag: n = lane&15, m = vgpr + 8*(lane>>4)
  #pragma unroll
  for (int nn = 0; nn < 4; ++nn)
    #pragma unroll
    for (int v = 0; v < 8; ++v) {
      long long gm = gm0 + wave * 16 + v + 8 * kh;
      long long gn = gn0 + nn * 16 + r;
      float val = alpha * acc[nn][v];
      if (bias)   val += bias[gm];
      if (doRelu) val = fmaxf(val, 0.0f);
      if (maskN)  val *= maskN[(long long)z * mZ + gn];
      long long off = cBase + (long long)z * sCz + gm * sm + gn * sn;
      if (outBf) ((u16*)Cv)[off] = f2b(val);
      else       ((float*)Cv)[off] = val;
    }
}

// ---------------------------------------------------------------------------
// Small kernels
// ---------------------------------------------------------------------------
__global__ void cvt_bf16_kernel(const float* __restrict__ src, u16* __restrict__ dst,
                                long long n) {
  long long i = (long long)blockIdx.x * blockDim.x + threadIdx.x;
  if (i < n) dst[i] = f2b(src[i]);
}

// W [O][I][3] f32  ->  Wp [3][O][I] bf16
__global__ void pack_conv_kernel(const float* __restrict__ src, u16* __restrict__ dst,
                                 int O, int I) {
  long long n = (long long)O * I * 3;
  long long i = (long long)blockIdx.x * blockDim.x + threadIdx.x;
  if (i >= n) return;
  long long o = i / ((long long)I * 3);
  long long rr = i % ((long long)I * 3);
  long long c = rr / 3, k = rr % 3;
  dst[(k * O + o) * I + c] = f2b(src[i]);
}

// x0 = x*mask (f32) and padded bf16 copy
__global__ void mask_cvt_kernel(const float* __restrict__ x, const float* __restrict__ mask,
                                float* __restrict__ xf, u16* __restrict__ xpad) {
  long long i = (long long)blockIdx.x * blockDim.x + threadIdx.x;
  if (i >= (long long)B_ * C_ * T_) return;
  int t = (int)(i % T_);
  long long bc = i / T_;
  int b = (int)(bc / C_);
  float val = x[i] * mask[b * T_ + t];
  xf[i] = val;
  xpad[bc * TP_ + t + 1] = f2b(val);
}

// scores[h][t][t+d-4] += SCALE * dot(q[:,t], rk[d])     (9-wide rel-k band)
__global__ void band_k_kernel(float* __restrict__ scores, const u16* __restrict__ q,
                              const float* __restrict__ rk) {
  int idx = blockIdx.x * blockDim.x + threadIdx.x;
  if (idx >= H_ * T_ * 9) return;
  int h = idx / (T_ * 9);
  int rr = idx % (T_ * 9);
  int t = rr / 9, d = rr % 9;
  int s = t + d - WIN_;
  if (s < 0 || s >= T_) return;
  const u16* qc = q + (long long)h * DK_ * T_ + t;
  const float* rkd = rk + d * DK_;
  float sum = 0.f;
  #pragma unroll 8
  for (int dk = 0; dk < DK_; ++dk) sum += b2f(qc[(long long)dk * T_]) * rkd[dk];
  scores[((long long)h * T_ + t) * T_ + s] += SCALE_ * sum;
}

// masked softmax over rows of [H*T][T]; writes bf16 P. 1 wave per row.
__global__ __launch_bounds__(128) void softmax_kernel(
    const float* __restrict__ scores, u16* __restrict__ P,
    const float* __restrict__ mask, int b) {
  const int lane = threadIdx.x & 31;
  const int wave = threadIdx.x >> 5;
  const int row  = blockIdx.x * 4 + wave;
  const int t = row % T_;
  const float maskT = mask[b * T_ + t];
  const float* src = scores + (long long)row * T_;
  float v[16];
  float mx = -3.0e38f;
  #pragma unroll
  for (int i = 0; i < 16; ++i) {
    int s = lane + 32 * i;
    float val = src[s];
    if (maskT * mask[b * T_ + s] == 0.f) val = -1e4f;
    v[i] = val;
    mx = fmaxf(mx, val);
  }
  for (int o = 16; o > 0; o >>= 1) mx = fmaxf(mx, __shfl_xor(mx, o, 32));
  float sum = 0.f;
  #pragma unroll
  for (int i = 0; i < 16; ++i) { v[i] = __expf(v[i] - mx); sum += v[i]; }
  for (int o = 16; o > 0; o >>= 1) sum += __shfl_xor(sum, o, 32);
  float inv = 1.0f / sum;
  u16* dst = P + (long long)row * T_;
  #pragma unroll
  for (int i = 0; i < 16; ++i) dst[lane + 32 * i] = f2b(v[i] * inv);
}

// attn[h*64+dk][t] += sum_d P[h][t][t+d-4] * rv[d][dk]   (rel-v band)
__global__ void band_v_kernel(u16* __restrict__ attn, const u16* __restrict__ P,
                              const float* __restrict__ rv) {
  int idx = blockIdx.x * blockDim.x + threadIdx.x;
  if (idx >= H_ * T_ * DK_) return;
  int h = idx / (T_ * DK_);
  int rr = idx % (T_ * DK_);
  int t = rr / DK_, dk = rr % DK_;
  const u16* prow = P + ((long long)h * T_ + t) * T_;
  float sum = 0.f;
  #pragma unroll
  for (int d = 0; d < 9; ++d) {
    int s = t + d - WIN_;
    if (s < 0 || s >= T_) continue;
    sum += b2f(prow[s]) * rv[d * DK_ + dk];
  }
  long long off = ((long long)h * DK_ + dk) * T_ + t;
  attn[off] = f2b(b2f(attn[off]) + sum);
}

// x = LN(x + y) over C; writes x (f32) and masked padded bf16 activation
__global__ __launch_bounds__(256) void resid_ln_kernel(
    float* __restrict__ x, const float* __restrict__ y,
    const float* __restrict__ g, const float* __restrict__ be,
    const float* __restrict__ mask, u16* __restrict__ xpad) {
  __shared__ float red[256];
  const int bt = blockIdx.x;
  const int b = bt / T_, t = bt % T_;
  const int tid = threadIdx.x;
  float v[3];
  float s = 0.f;
  #pragma unroll
  for (int i = 0; i < 3; ++i) {
    int c = tid + i * 256;
    long long off = ((long long)(b * C_ + c)) * T_ + t;
    float val = x[off] + y[off];
    v[i] = val; s += val;
  }
  red[tid] = s; __syncthreads();
  for (int o = 128; o > 0; o >>= 1) { if (tid < o) red[tid] += red[tid + o]; __syncthreads(); }
  float mean = red[0] / (float)C_;
  __syncthreads();
  float s2 = 0.f;
  #pragma unroll
  for (int i = 0; i < 3; ++i) { float d = v[i] - mean; s2 += d * d; }
  red[tid] = s2; __syncthreads();
  for (int o = 128; o > 0; o >>= 1) { if (tid < o) red[tid] += red[tid + o]; __syncthreads(); }
  float rinv = rsqrtf(red[0] / (float)C_ + 1e-5f);
  float mk = mask[b * T_ + t];
  #pragma unroll
  for (int i = 0; i < 3; ++i) {
    int c = tid + i * 256;
    long long off = ((long long)(b * C_ + c)) * T_ + t;
    float o = (v[i] - mean) * rinv * g[c] + be[c];
    x[off] = o;
    xpad[((long long)(b * C_ + c)) * TP_ + t + 1] = f2b(o * mk);
  }
}

__global__ void final_kernel(const float* __restrict__ xf, const float* __restrict__ mask,
                             float* __restrict__ out) {
  long long i = (long long)blockIdx.x * blockDim.x + threadIdx.x;
  if (i >= (long long)B_ * C_ * T_) return;
  int t = (int)(i % T_);
  int b = (int)(i / ((long long)C_ * T_));
  out[i] = xf[i] * mask[b * T_ + t];
}

// ---------------------------------------------------------------------------
extern "C" void kernel_launch(void* const* d_in, const int* in_sizes, int n_in,
                              void* d_out, int out_size, void* d_ws, size_t ws_size,
                              hipStream_t stream) {
  (void)in_sizes; (void)n_in; (void)out_size; (void)ws_size;
  const float* x_in = (const float*)d_in[0];
  const float* mask = (const float*)d_in[1];
  const float* Wq = (const float*)d_in[2];   const float* bq = (const float*)d_in[3];
  const float* Wk = (const float*)d_in[4];   const float* bk = (const float*)d_in[5];
  const float* Wv = (const float*)d_in[6];   const float* bv = (const float*)d_in[7];
  const float* Wo = (const float*)d_in[8];   const float* bo = (const float*)d_in[9];
  const float* rk = (const float*)d_in[10];  const float* rv = (const float*)d_in[11];
  const float* W1 = (const float*)d_in[12];  const float* b1 = (const float*)d_in[13];
  const float* W2 = (const float*)d_in[14];  const float* b2 = (const float*)d_in[15];
  const float* g1 = (const float*)d_in[16];  const float* be1 = (const float*)d_in[17];
  const float* g2 = (const float*)d_in[18];  const float* be2 = (const float*)d_in[19];

  char* wsp = (char*)d_ws;
  auto alloc = [&](size_t bytes) -> char* {
    char* p = wsp; wsp += (bytes + 255) & ~(size_t)255; return p;
  };
  const size_t NBT = (size_t)B_ * C_ * T_;
  float* xf   = (float*)alloc(NBT * 4);
  float* ybuf = (float*)alloc(NBT * 4);
  u16* xpad   = (u16*)alloc((size_t)B_ * C_ * TP_ * 2);
  u16* qbuf   = (u16*)alloc(NBT * 2);
  u16* kbuf   = (u16*)alloc(NBT * 2);
  u16* vbuf   = (u16*)alloc(NBT * 2);
  u16* attn   = (u16*)alloc(NBT * 2);
  float* sc   = (float*)alloc((size_t)H_ * T_ * T_ * 4);
  u16* Pb     = (u16*)alloc((size_t)H_ * T_ * T_ * 2);
  u16* hpad   = (u16*)alloc((size_t)B_ * FC_ * TP_ * 2);
  u16* wqp    = (u16*)alloc((size_t)C_ * C_ * 2);
  u16* wkp    = (u16*)alloc((size_t)C_ * C_ * 2);
  u16* wvp    = (u16*)alloc((size_t)C_ * C_ * 2);
  u16* wop    = (u16*)alloc((size_t)C_ * C_ * 2);
  u16* w1p    = (u16*)alloc((size_t)3 * FC_ * C_ * 2);
  u16* w2p    = (u16*)alloc((size_t)3 * C_ * FC_ * 2);

  (void)hipMemsetAsync(xpad, 0, (size_t)B_ * C_ * TP_ * 2, stream);
  (void)hipMemsetAsync(hpad, 0, (size_t)B_ * FC_ * TP_ * 2, stream);

  {
    long long n = (long long)NBT;
    mask_cvt_kernel<<<dim3((unsigned)((n + 255) / 256)), 256, 0, stream>>>(x_in, mask, xf, xpad);
  }

  auto GEMM = [&](int tA, int tB, long long M, long long N,
                  const u16* A, long long lda, long long sAz, long long sAkb,
                  const u16* Bp, long long ldb, long long sBz, long long sBkb,
                  void* Cp, long long cBase, long long sCz, long long smv, long long snv,
                  int outBf, const float* bias, const float* mN, long long mZ,
                  int Kd, int nkb, float alpha, int relu, int nz) {
    dim3 grid((unsigned)(M / 128), (unsigned)(N / 64), (unsigned)nz);
    if (tA == 0 && tB == 0)
      gemm_bf16_wmma<0, 0><<<grid, 256, 0, stream>>>(A, lda, sAz, sAkb, Bp, ldb, sBz, sBkb,
          Cp, cBase, sCz, smv, snv, outBf, bias, mN, mZ, Kd, nkb, alpha, relu);
    else if (tA == 1 && tB == 0)
      gemm_bf16_wmma<1, 0><<<grid, 256, 0, stream>>>(A, lda, sAz, sAkb, Bp, ldb, sBz, sBkb,
          Cp, cBase, sCz, smv, snv, outBf, bias, mN, mZ, Kd, nkb, alpha, relu);
    else
      gemm_bf16_wmma<0, 1><<<grid, 256, 0, stream>>>(A, lda, sAz, sAkb, Bp, ldb, sBz, sBkb,
          Cp, cBase, sCz, smv, snv, outBf, bias, mN, mZ, Kd, nkb, alpha, relu);
  };

  const long long CT  = (long long)C_ * T_;
  const long long CTP = (long long)C_ * TP_;

  for (int l = 0; l < L_; ++l) {
    // ---- pack this layer's weights to bf16 ----
    long long ncc = (long long)C_ * C_;
    dim3 gcc((unsigned)((ncc + 255) / 256));
    cvt_bf16_kernel<<<gcc, 256, 0, stream>>>(Wq + (size_t)l * ncc, wqp, ncc);
    cvt_bf16_kernel<<<gcc, 256, 0, stream>>>(Wk + (size_t)l * ncc, wkp, ncc);
    cvt_bf16_kernel<<<gcc, 256, 0, stream>>>(Wv + (size_t)l * ncc, wvp, ncc);
    cvt_bf16_kernel<<<gcc, 256, 0, stream>>>(Wo + (size_t)l * ncc, wop, ncc);
    long long n1 = (long long)FC_ * C_ * 3;
    pack_conv_kernel<<<dim3((unsigned)((n1 + 255) / 256)), 256, 0, stream>>>(
        W1 + (size_t)l * n1, w1p, FC_, C_);
    pack_conv_kernel<<<dim3((unsigned)((n1 + 255) / 256)), 256, 0, stream>>>(
        W2 + (size_t)l * n1, w2p, C_, FC_);

    // ---- Q,K,V projections:  W[CxC] * xpad[CxT]  (z = batch) ----
    GEMM(0, 0, C_, T_, wqp, C_, 0, 0, xpad + 1, TP_, CTP, 0,
         qbuf, 0, CT, T_, 1, 1, bq + (size_t)l * C_, nullptr, 0, C_, 1, 1.f, 0, B_);
    GEMM(0, 0, C_, T_, wkp, C_, 0, 0, xpad + 1, TP_, CTP, 0,
         kbuf, 0, CT, T_, 1, 1, bk + (size_t)l * C_, nullptr, 0, C_, 1, 1.f, 0, B_);
    GEMM(0, 0, C_, T_, wvp, C_, 0, 0, xpad + 1, TP_, CTP, 0,
         vbuf, 0, CT, T_, 1, 1, bv + (size_t)l * C_, nullptr, 0, C_, 1, 1.f, 0, B_);

    // ---- attention, per batch (z = head) ----
    for (int b = 0; b < B_; ++b) {
      long long bOff = (long long)b * CT;
      // scores = SCALE * Q^T K   (A = q slice stored [dk x t] -> TA=1)
      GEMM(1, 0, T_, T_, qbuf + bOff, T_, (long long)DK_ * T_, 0,
           kbuf + bOff, T_, (long long)DK_ * T_, 0,
           sc, 0, (long long)T_ * T_, T_, 1, 0, nullptr, nullptr, 0,
           DK_, 1, SCALE_, 0, H_);
      band_k_kernel<<<dim3((H_ * T_ * 9 + 255) / 256), 256, 0, stream>>>(sc, qbuf + bOff, rk);
      softmax_kernel<<<dim3(H_ * T_ / 4), 128, 0, stream>>>(sc, Pb, mask, b);
      // out = P * V^T  (B = v slice stored [dk x s] -> TB=1); write [h*64+dk][t]
      GEMM(0, 1, T_, DK_, Pb, T_, (long long)T_ * T_, 0,
           vbuf + bOff, T_, (long long)DK_ * T_, 0,
           attn, bOff, (long long)DK_ * T_, 1, T_, 1, nullptr, nullptr, 0,
           T_, 1, 1.f, 0, H_);
      band_v_kernel<<<dim3((H_ * T_ * DK_ + 255) / 256), 256, 0, stream>>>(attn + bOff, Pb, rv);
    }

    // ---- output projection (f32 result) ----
    GEMM(0, 0, C_, T_, wop, C_, 0, 0, attn, T_, CT, 0,
         ybuf, 0, CT, T_, 1, 0, bo + (size_t)l * C_, nullptr, 0, C_, 1, 1.f, 0, B_);
    resid_ln_kernel<<<dim3(B_ * T_), 256, 0, stream>>>(
        xf, ybuf, g1 + (size_t)l * C_, be1 + (size_t)l * C_, mask, xpad);

    // ---- conv FFN: K=3 conv as 3 shifted GEMMs (nkb=3, B stride = 1 col) ----
    GEMM(0, 0, FC_, T_, w1p, C_, 0, (long long)FC_ * C_,
         xpad, TP_, CTP, 1,
         hpad, 1, (long long)FC_ * TP_, TP_, 1, 1,
         b1 + (size_t)l * FC_, mask, T_, C_, 3, 1.f, 1, B_);
    GEMM(0, 0, C_, T_, w2p, FC_, 0, (long long)C_ * FC_,
         hpad, TP_, (long long)FC_ * TP_, 1,
         ybuf, 0, CT, T_, 1, 0,
         b2 + (size_t)l * C_, mask, T_, FC_, 3, 1.f, 0, B_);
    resid_ln_kernel<<<dim3(B_ * T_), 256, 0, stream>>>(
        xf, ybuf, g2 + (size_t)l * C_, be2 + (size_t)l * C_, mask, xpad);
  }

  final_kernel<<<dim3((unsigned)((NBT + 255) / 256)), 256, 0, stream>>>(xf, mask, (float*)d_out);
}
